// STSSTransformation_51299089383988
// MI455X (gfx1250) — matrix operand order repaired and checked by reading, not compiled, not run on hardware
//
#include <hip/hip_runtime.h>
#include <hip/hip_bf16.h>

typedef __attribute__((ext_vector_type(16))) _Float16 v16h;
typedef __attribute__((ext_vector_type(8)))  _Float16 v8h;
typedef __attribute__((ext_vector_type(8)))  float    v8f;

#define NFRAMES 64      // b*T
#define C       384     // channels
#define HW      196     // 14*14
#define NP      208     // padded positions (13 tiles * 16)
#define NTILE   13
#define TT      8       // T
#define LL      5       // temporal window
#define WIN     9       // spatial window

// ---------------------------------------------------------------------------
// Phase 1: L2-normalize over channels, emit f16 in channel-contiguous layout
//   fN[frame][p][c],  p padded to 208 with zeros.
// One wave (32 lanes) per (frame, padded position). Each lane owns 12 channels.
// ---------------------------------------------------------------------------
__global__ __launch_bounds__(256)
void stss_normalize_kernel(const float* __restrict__ x, _Float16* __restrict__ fN) {
    const int wave = blockIdx.x * (blockDim.x >> 5) + (threadIdx.x >> 5);
    const int lane = threadIdx.x & 31;
    if (wave >= NFRAMES * NP) return;
    const int frame = wave / NP;
    const int p     = wave % NP;

    _Float16* outp = fN + ((size_t)frame * NP + p) * C;

    if (p >= HW) {                       // zero the tile padding rows
        for (int i = lane; i < C; i += 32) outp[i] = (_Float16)0.0f;
        return;
    }

    const float* xp = x + (size_t)frame * C * HW + p;   // stride HW between channels
    float vals[12];
    float ss = 0.0f;
#pragma unroll
    for (int i = 0; i < 12; ++i) {
        float v = xp[(size_t)(lane + 32 * i) * HW];
        vals[i] = v;
        ss += v * v;
    }
    // wave32 butterfly reduction
#pragma unroll
    for (int off = 16; off >= 1; off >>= 1) ss += __shfl_xor(ss, off, 32);

    const float inv = 1.0f / fmaxf(sqrtf(ss), 1e-12f);
#pragma unroll
    for (int i = 0; i < 12; ++i)
        outp[lane + 32 * i] = (_Float16)(vals[i] * inv);
}

// ---------------------------------------------------------------------------
// Phase 2: banded Gram matrix via WMMA + windowed scatter.
// grid = (13, 13, b*T*L); one wave per 16x16 tile of C = f1^T * f2 (196x196,
// K=384). Skips tile pairs outside the |q-p|<=60 band and invalid frames.
//
// gfx1250 WMMA f16 16x16x32 fragments as loaded here (contiguous k, same for A and B):
//   A (16x32, MxK): lane l -> row M=l%16; elems[0..15]=K=16*(l/16)+0..15
//   B (32x16, KxN): lane l -> col N=l%16; elems[0..15]=K=16*(l/16)+0..15
//   C (16x16 f32) : lane l, vgpr r -> M=r+8*(l/16), N=l%16
// With fN channel-contiguous, A = one 32B load, B = one 32B load per lane.
// ---------------------------------------------------------------------------
__global__ __launch_bounds__(32)
void stss_corr_kernel(const _Float16* __restrict__ fN, float* __restrict__ out) {
    const int ti = blockIdx.x;            // row tile (f1 positions)
    const int tj = blockIdx.y;            // col tile (f2 positions)
    const int z  = blockIdx.z;            // (b, t, li)
    const int li = z % LL;
    const int t  = (z / LL) % TT;
    const int b  = z / (LL * TT);

    const int t2 = t + li - (LL / 2);
    if (t2 < 0 || t2 >= TT) return;       // zero slice already from memset

    const int p0 = ti * 16, q0 = tj * 16;
    const int d  = q0 - p0;               // band: |q-p| <= 60 possible iff |d| <= 75
    if (d > 75 || d < -75) return;

    const int lane = threadIdx.x;
    const int half = lane >> 4;           // 0: lanes 0-15, 1: lanes 16-31
    const int lm   = lane & 15;

    const _Float16* Arow = fN + ((size_t)(b * TT + t)  * NP + (p0 + lm)) * C;
    const _Float16* Bcol = fN + ((size_t)(b * TT + t2) * NP + (q0 + lm)) * C;

    v8f acc = {};
#pragma unroll
    for (int k0 = 0; k0 < C; k0 += 32) {
        v16h a  = *(const v16h*)(Arow + k0 + 16 * half);
        v16h bm = *(const v16h*)(Bcol + k0 + 16 * half);
        acc = __builtin_amdgcn_wmma_f32_16x16x32_f16(
                  /*neg_a=*/false, a, /*neg_b=*/false, bm,
                  /*c_mod=*/(short)0, acc, /*reuse_a=*/false, /*reuse_b=*/false);
    }

    // Scatter this lane's column of the Gram tile into the (u,v) window slots.
    const int q = q0 + lm;                // f2 position owned by this lane
    if (q >= HW) return;
    const int qh = q / 14, qw = q % 14;
    const size_t frameBase = (size_t)(b * TT + t);
#pragma unroll
    for (int r = 0; r < 8; ++r) {
        const int m = r + 8 * half;       // C-matrix row within tile
        const int p = p0 + m;             // f1 position
        if (p >= HW) continue;
        const int h = p / 14, w = p % 14;
        const int dh = qh - h, dw = qw - w;
        if (dh < -4 || dh > 4 || dw < -4 || dw > 4) continue;
        const size_t oidx =
            ((((frameBase * 14 + h) * 14 + w) * LL + li) * (WIN * WIN))
            + (size_t)(dh + 4) * WIN + (dw + 4);
        out[oidx] = acc[r];
    }
}

extern "C" void kernel_launch(void* const* d_in, const int* in_sizes, int n_in,
                              void* d_out, int out_size, void* d_ws, size_t ws_size,
                              hipStream_t stream) {
    const float* x  = (const float*)d_in[0];
    float*       out = (float*)d_out;
    _Float16*    fN  = (_Float16*)d_ws;   // 64*208*384*2 = ~9.75 MB

    // Spatially-OOB window slots and temporally-OOB frames are exactly zero.
    hipMemsetAsync(d_out, 0, (size_t)out_size * sizeof(float), stream);

    // Phase 1: normalize + f16 repack (one wave per padded position)
    {
        const int waves  = NFRAMES * NP;           // 13312
        const int block  = 256;                    // 8 waves/block
        const int blocks = (waves + 7) / 8;
        stss_normalize_kernel<<<blocks, block, 0, stream>>>(x, fN);
    }

    // Phase 2: banded WMMA Gram + scatter
    {
        dim3 grid(NTILE, NTILE, 8 * TT * LL);      // (13, 13, 320)
        stss_corr_kernel<<<grid, 32, 0, stream>>>(fN, out);
    }
}
